// rt_GCN_Geo_65979287601512
// MI455X (gfx1250) — matrix operand order repaired and checked by reading, not compiled, not run on hardware
//
#include <hip/hip_runtime.h>
#include <hip/hip_bf16.h>

// ---------------------------------------------------------------------------
// GNN pipeline for MI455X (gfx1250, wave32).
//  * NNConv refactored: Z = X @ [W_0..W_10 | nn_b | root] (one WMMA GEMM per
//    layer), then a cheap per-(edge,output) contraction + f32 atomics.
//  * All dense GEMMs: operands pre-packed to padded bf16 (A row-major, B
//    transposed) so the WMMA kernel is branchless: 4x b128 loads + 1 wmma per
//    16x16x32 step, fully unrolled over K (templated KPAD), templated epilogue.
//  * Everything fits in the 192MB L2; workload is launch/latency bound.
// ---------------------------------------------------------------------------

#define N_ATOMS 100000
#define E_ATOMS 200000
#define N_AMINO 10240
#define E_AMINO 20480
#define BGRAPH  512
#define D_IN    40
#define D_E     11
#define HD      20                 // H1 == H2 == H3
#define CCAT    ((D_E + 2) * HD)   // 260 fused NNConv output channels
#define ZLD     272                // CCAT padded to multiple of 16
#define D_AA    8
#define F_AIN   (HD + D_AA)        // 28
#define H_ARMA  128
#define KST     3
#define TT      7
#define F1      256
#define F2      128
#define F3      64

typedef __bf16 bf16x8  __attribute__((ext_vector_type(8)));
typedef __bf16 bf16x16 __attribute__((ext_vector_type(16)));
typedef float  f32x8   __attribute__((ext_vector_type(8)));

static inline int ceilDiv(int a, int b) { return (a + b - 1) / b; }

// ---------------------------------------------------------------------------
// Branchless WMMA GEMM on pre-packed operands.
//   Ap : bf16 [M x KPAD] row-major (K zero-padded)
//   Bt : bf16 [Npad x KPAD] row-major = B^T (N,K zero-padded)
//   C  : f32  [M x ldc], stores unconditional (Npad == grid.x*16 <= ldc)
// M must be a multiple of 16. One wave per 16x16 tile, 4 waves per block.
// EPI: 0 = plain store, 1 = +bias, 2 = +bias,ReLU (compile-time).
// ---------------------------------------------------------------------------
template <int KPAD, int EPI>
__global__ void gemm_bf16_wmma(const __bf16* __restrict__ Ap,
                               const __bf16* __restrict__ Bt,
                               float* __restrict__ C, int ldc, int M,
                               const float* __restrict__ bias)
{
    const int lane  = threadIdx.x;                        // 0..31
    const int tileN = blockIdx.x;
    const int tileM = blockIdx.y * blockDim.y + threadIdx.y;
    if (tileM * 16 >= M) return;                          // uniform per wave

    const int r  = lane & 15;
    const int hi = lane >> 4;
    const __bf16* aptr = Ap + (size_t)(tileM * 16 + r) * KPAD;
    const __bf16* bptr = Bt + (size_t)(tileN * 16 + r) * KPAD;

    f32x8 acc = {0.f, 0.f, 0.f, 0.f, 0.f, 0.f, 0.f, 0.f};

    #pragma unroll
    for (int k0 = 0; k0 < KPAD; k0 += 32) {
        // A fragment (ISA 7.12.2 16x32 layout): lane half selects K 0..7/16..23
        // vs 8..15/24..31 — two contiguous 16B loads.
        bf16x8 a0 = *(const bf16x8*)(aptr + k0 + hi * 8);
        bf16x8 a1 = *(const bf16x8*)(aptr + k0 + 16 + hi * 8);
        bf16x16 afr = __builtin_shufflevector(a0, a1, 0, 1, 2, 3, 4, 5, 6, 7,
                                              8, 9, 10, 11, 12, 13, 14, 15);
        // B fragment: per-lane column, same K split as A (0..7/16..23 vs 8..15/24..31).
        bf16x8 b0 = *(const bf16x8*)(bptr + k0 + hi * 8);
        bf16x8 b1 = *(const bf16x8*)(bptr + k0 + 16 + hi * 8);
        bf16x16 bfr = __builtin_shufflevector(b0, b1, 0, 1, 2, 3, 4, 5, 6, 7,
                                              8, 9, 10, 11, 12, 13, 14, 15);
        acc = __builtin_amdgcn_wmma_f32_16x16x32_bf16(
                  false, afr, false, bfr, (short)0, acc, false, false);
    }

    float bb = 0.0f;
    if (EPI >= 1) bb = bias[tileN * 16 + r];              // one load per lane
    float* cptr = C + (size_t)(tileM * 16 + hi * 8) * ldc + tileN * 16 + r;
    #pragma unroll
    for (int v = 0; v < 8; ++v) {
        float o = acc[v];
        if (EPI >= 1) o += bb;
        if (EPI == 2) o = fmaxf(o, 0.0f);
        cptr[(size_t)v * ldc] = o;
    }
}

// A (f32, M x K, leading dim lda) -> Ap (bf16, M x Kpad, zero-padded K)
__global__ void packA(const float* __restrict__ A, __bf16* __restrict__ Ap,
                      int M, int K, int lda, int Kpad)
{
    int idx = blockIdx.x * blockDim.x + threadIdx.x;
    if (idx >= M * Kpad) return;
    int m = idx / Kpad, k = idx % Kpad;
    Ap[idx] = (__bf16)((k < K) ? A[(size_t)m * lda + k] : 0.0f);
}

// B (f32, K x N, leading dim ldb) -> Bt (bf16, Npad x Kpad, transposed+padded)
__global__ void packT(const float* __restrict__ B, __bf16* __restrict__ Bt,
                      int K, int N, int ldb, int Kpad, int Npad)
{
    int idx = blockIdx.x * blockDim.x + threadIdx.x;
    if (idx >= Npad * Kpad) return;
    int n = idx / Kpad, k = idx % Kpad;
    Bt[idx] = (__bf16)((n < N && k < K) ? B[(size_t)k * ldb + n] : 0.0f);
}

// Build Wcat^T directly as padded bf16: rows = fused channels (272), cols = Kpad
// channel c: k=c/20 -> W_k (k<11), nn_b matrix (k==11), root (k==12)
__global__ void build_wcatT(const float* __restrict__ nnw,   // [D_E, di*HD]
                            const float* __restrict__ nnb,   // [di*HD]
                            const float* __restrict__ root,  // [di, HD]
                            __bf16* __restrict__ Bt, int di, int Kpad)
{
    int idx = blockIdx.x * blockDim.x + threadIdx.x;
    if (idx >= ZLD * Kpad) return;
    int c = idx / Kpad, i = idx % Kpad;
    float v = 0.0f;
    if (c < CCAT && i < di) {
        int k = c / HD, o = c % HD;
        if (k < D_E)       v = nnw[(size_t)k * (di * HD) + i * HD + o];
        else if (k == D_E) v = nnb[i * HD + o];
        else               v = root[i * HD + o];
    }
    Bt[idx] = (__bf16)v;
}

// ---------------------------------------------------------------------------
__global__ void zero_f32(float* __restrict__ p, long long n)
{
    long long i = (long long)blockIdx.x * blockDim.x + threadIdx.x;
    long long stride = (long long)gridDim.x * blockDim.x;
    for (; i < n; i += stride) p[i] = 0.0f;
}

// Per-(edge,output) message: msg = Z[s][11*20+o] + sum_k ea[e,k] * Z[s][k*20+o]
// Lanes cover consecutive o of one edge -> coalesced Z reads and agg atomics.
__global__ void nnconv_edge(const int* __restrict__ ei,
                            const float* __restrict__ ea,
                            const float* __restrict__ Z,
                            float* __restrict__ agg)
{
    int idx = blockIdx.x * blockDim.x + threadIdx.x;
    if (idx >= E_ATOMS * HD) return;
    int e = idx / HD, o = idx % HD;
    int s = ei[e];
    int d = ei[E_ATOMS + e];
    const float* zs = Z + (size_t)s * ZLD;
    float m = zs[D_E * HD + o];
    #pragma unroll
    for (int k = 0; k < D_E; ++k)
        m = fmaf(ea[(size_t)e * D_E + k], zs[k * HD + o], m);
    atomicAdd(agg + (size_t)d * HD + o, m);
}

// h = relu(root_part + agg + bias)
__global__ void nnconv_finalize(const float* __restrict__ Z,
                                const float* __restrict__ agg,
                                const float* __restrict__ bias,
                                float* __restrict__ h)
{
    int idx = blockIdx.x * blockDim.x + threadIdx.x;
    if (idx >= N_ATOMS * HD) return;
    int i = idx / HD, o = idx % HD;
    float v = Z[(size_t)i * ZLD + (D_E + 1) * HD + o] + agg[idx] + bias[o];
    h[idx] = fmaxf(v, 0.0f);
}

// aa[:, :20] = 0 (scatter target), aa[:, 20:28] = amino features
__global__ void aa_init(const float* __restrict__ feat, float* __restrict__ aa)
{
    int idx = blockIdx.x * blockDim.x + threadIdx.x;
    if (idx >= N_AMINO * F_AIN) return;
    int n = idx / F_AIN, o = idx % F_AIN;
    aa[idx] = (o < HD) ? 0.0f : feat[(size_t)n * D_AA + (o - HD)];
}

__global__ void aa_scatter(const int* __restrict__ labels,
                           const float* __restrict__ h, float* __restrict__ aa)
{
    int idx = blockIdx.x * blockDim.x + threadIdx.x;
    if (idx >= N_ATOMS * HD) return;
    int n = idx / HD, o = idx % HD;
    atomicAdd(aa + (size_t)labels[n] * F_AIN + o, h[idx]);
}

__global__ void deg_scatter(const int* __restrict__ aei, float* __restrict__ deg)
{
    int e = blockIdx.x * blockDim.x + threadIdx.x;
    if (e >= E_AMINO) return;
    atomicAdd(deg + aei[E_AMINO + e], 1.0f);
}

__global__ void deg_to_dinv(float* __restrict__ deg)
{
    int n = blockIdx.x * blockDim.x + threadIdx.x;
    if (n >= N_AMINO) return;
    float d = deg[n];
    deg[n] = (d > 0.0f) ? rsqrtf(d) : 0.0f;
}

__global__ void edge_norm(const int* __restrict__ aei,
                          const float* __restrict__ dinv, float* __restrict__ nrm)
{
    int e = blockIdx.x * blockDim.x + threadIdx.x;
    if (e >= E_AMINO) return;
    nrm[e] = dinv[aei[e]] * dinv[aei[E_AMINO + e]];
}

// acc[k][dst] += norm[e] * tmp[k][src]   (all K stacks in one launch)
__global__ void arma_scatter(const int* __restrict__ aei,
                             const float* __restrict__ nrm,
                             const float* __restrict__ tmp,
                             float* __restrict__ acc)
{
    long long idx = (long long)blockIdx.x * blockDim.x + threadIdx.x;
    if (idx >= (long long)KST * E_AMINO * H_ARMA) return;
    int o = (int)(idx & (H_ARMA - 1));
    long long rr = idx >> 7;
    int e = (int)(rr % E_AMINO);
    int k = (int)(rr / E_AMINO);
    int s = aei[e];
    int d = aei[E_AMINO + e];
    float v = nrm[e] * tmp[((size_t)k * N_AMINO + s) * H_ARMA + o];
    atomicAdd(acc + ((size_t)k * N_AMINO + d) * H_ARMA + o, v);
}

__global__ void relu_copy(const float* __restrict__ src, float* __restrict__ dst,
                          long long n)
{
    long long i = (long long)blockIdx.x * blockDim.x + threadIdx.x;
    if (i >= n) return;
    dst[i] = fmaxf(src[i], 0.0f);
}

// aaout = relu(mean over K stacks)
__global__ void arma_mean_relu(const float* __restrict__ prev,
                               float* __restrict__ aaout)
{
    int idx = blockIdx.x * blockDim.x + threadIdx.x;
    if (idx >= N_AMINO * H_ARMA) return;
    const long long NH = (long long)N_AMINO * H_ARMA;
    float v = (prev[idx] + prev[idx + NH] + prev[idx + 2 * NH]) * (1.0f / 3.0f);
    aaout[idx] = fmaxf(v, 0.0f);
}

__global__ void pool_graph(const int* __restrict__ batch,
                           const float* __restrict__ aaout, float* __restrict__ g)
{
    int idx = blockIdx.x * blockDim.x + threadIdx.x;
    if (idx >= N_AMINO * H_ARMA) return;
    int n = idx >> 7, o = idx & (H_ARMA - 1);
    atomicAdd(g + (size_t)batch[n] * H_ARMA + o, aaout[idx]);
}

__global__ void final_dot(const float* __restrict__ p3,
                          const float* __restrict__ w,
                          const float* __restrict__ b, float* __restrict__ out)
{
    int i = blockIdx.x * blockDim.x + threadIdx.x;
    if (i >= BGRAPH) return;
    float s = b[0];
    #pragma unroll
    for (int f = 0; f < F3; ++f) s = fmaf(p3[(size_t)i * F3 + f], w[f], s);
    out[i] = s;
}

// ---------------------------------------------------------------------------
template <int KPAD>
static void launch_gemm_k(dim3 grid, dim3 blk, hipStream_t s,
                          const __bf16* Ap, const __bf16* Bt, float* C, int ldc,
                          int M, const float* bias, int epi)
{
    if (epi == 0)
        gemm_bf16_wmma<KPAD, 0><<<grid, blk, 0, s>>>(Ap, Bt, C, ldc, M, nullptr);
    else if (epi == 1)
        gemm_bf16_wmma<KPAD, 1><<<grid, blk, 0, s>>>(Ap, Bt, C, ldc, M, bias);
    else
        gemm_bf16_wmma<KPAD, 2><<<grid, blk, 0, s>>>(Ap, Bt, C, ldc, M, bias);
}

static void launch_gemm(int Kpad, dim3 grid, dim3 blk, hipStream_t s,
                        const __bf16* Ap, const __bf16* Bt, float* C, int ldc,
                        int M, const float* bias, int epi)
{
    switch (Kpad) {
    case 32:  launch_gemm_k<32>(grid, blk, s, Ap, Bt, C, ldc, M, bias, epi); break;
    case 64:  launch_gemm_k<64>(grid, blk, s, Ap, Bt, C, ldc, M, bias, epi); break;
    case 128: launch_gemm_k<128>(grid, blk, s, Ap, Bt, C, ldc, M, bias, epi); break;
    case 256: launch_gemm_k<256>(grid, blk, s, Ap, Bt, C, ldc, M, bias, epi); break;
    default: break;
    }
}

extern "C" void kernel_launch(void* const* d_in, const int* in_sizes, int n_in,
                              void* d_out, int out_size, void* d_ws, size_t ws_size,
                              hipStream_t stream)
{
    const float* x      = (const float*)d_in[0];
    const int*   ei     = (const int*)d_in[1];
    const float* ea     = (const float*)d_in[2];
    const int*   labels = (const int*)d_in[3];
    const float* aaf    = (const float*)d_in[4];
    const int*   aei    = (const int*)d_in[5];
    const int*   abatch = (const int*)d_in[6];
    const float* nnw[3]  = {(const float*)d_in[7],  (const float*)d_in[11], (const float*)d_in[15]};
    const float* nnb[3]  = {(const float*)d_in[8],  (const float*)d_in[12], (const float*)d_in[16]};
    const float* root[3] = {(const float*)d_in[9],  (const float*)d_in[13], (const float*)d_in[17]};
    const float* lb[3]   = {(const float*)d_in[10], (const float*)d_in[14], (const float*)d_in[18]};
    const float* arma_init = (const float*)d_in[19];
    const float* arma_w    = (const float*)d_in[20];
    const float* arma_root = (const float*)d_in[21];
    const float* arma_bias = (const float*)d_in[22];
    const float* l1_w = (const float*)d_in[23]; const float* l1_b = (const float*)d_in[24];
    const float* l2_w = (const float*)d_in[25]; const float* l2_b = (const float*)d_in[26];
    const float* l3_w = (const float*)d_in[27]; const float* l3_b = (const float*)d_in[28];
    const float* l4_w = (const float*)d_in[29]; const float* l4_b = (const float*)d_in[30];
    float* out = (float*)d_out;

    // ---- workspace bump allocator (~200 MB, dominated by Z) ----
    char* cur = (char*)d_ws;
    auto allocf = [&](size_t n) -> float* {
        float* p = (float*)cur;
        cur += ((n * sizeof(float) + 255) & ~size_t(255));
        return p;
    };
    auto allocbf = [&](size_t n) -> __bf16* {
        __bf16* p = (__bf16*)cur;
        cur += ((n * sizeof(__bf16) + 255) & ~size_t(255));
        return p;
    };
    float*  Z     = allocf((size_t)N_ATOMS * ZLD);        // 108.8 MB
    float*  hA    = allocf((size_t)N_ATOMS * HD);
    float*  hB    = allocf((size_t)N_ATOMS * HD);
    float*  agg   = allocf((size_t)N_ATOMS * HD);
    float*  aa    = allocf((size_t)N_AMINO * F_AIN);
    float*  dinv  = allocf((size_t)N_AMINO);
    float*  nrm   = allocf((size_t)E_AMINO);
    float*  prev  = allocf((size_t)KST * N_AMINO * H_ARMA);
    float*  tmp   = allocf((size_t)KST * N_AMINO * H_ARMA);
    float*  acc   = allocf((size_t)KST * N_AMINO * H_ARMA);
    float*  aaout = allocf((size_t)N_AMINO * H_ARMA);
    float*  g     = allocf((size_t)BGRAPH * H_ARMA);
    float*  p1    = allocf((size_t)BGRAPH * F1);
    float*  p2    = allocf((size_t)BGRAPH * F2);
    float*  p3    = allocf((size_t)BGRAPH * F3);
    __bf16* Apack = allocbf((size_t)N_ATOMS * 64);        // 12.8 MB, reused by all GEMMs
    __bf16* aabf  = allocbf((size_t)N_AMINO * 32);        // packed aa, reused across T
    __bf16* btbuf = allocbf((size_t)32768);               // packed B^T, reused

    const dim3 wblk(32, 4);  // 4 waves / block, one 16x16 tile per wave

    // ================= 3 x NNConv (atom graph) =================
    const float* xin = x;
    int di = D_IN;
    float* hout = hA;
    for (int L = 0; L < 3; ++L) {
        int Kpad = (di + 31) & ~31;  // 64 then 32
        packA<<<ceilDiv(N_ATOMS * Kpad, 256), 256, 0, stream>>>(
            xin, Apack, N_ATOMS, di, di, Kpad);
        build_wcatT<<<ceilDiv(ZLD * Kpad, 256), 256, 0, stream>>>(
            nnw[L], nnb[L], root[L], btbuf, di, Kpad);
        dim3 gz(ZLD / 16, ceilDiv(N_ATOMS / 16, 4));
        launch_gemm(Kpad, gz, wblk, stream, Apack, btbuf, Z, ZLD, N_ATOMS,
                    nullptr, 0);
        zero_f32<<<2048, 256, 0, stream>>>(agg, (long long)N_ATOMS * HD);
        nnconv_edge<<<ceilDiv(E_ATOMS * HD, 256), 256, 0, stream>>>(ei, ea, Z, agg);
        nnconv_finalize<<<ceilDiv(N_ATOMS * HD, 256), 256, 0, stream>>>(
            Z, agg, lb[L], hout);
        xin = hout;
        di = HD;
        hout = (hout == hA) ? hB : hA;
    }
    const float* h3 = xin;

    // ======= atoms -> amino readout + feature concat =======
    aa_init<<<ceilDiv(N_AMINO * F_AIN, 256), 256, 0, stream>>>(aaf, aa);
    aa_scatter<<<ceilDiv(N_ATOMS * HD, 256), 256, 0, stream>>>(labels, h3, aa);

    // ======= symmetric normalization for ARMA =======
    zero_f32<<<64, 256, 0, stream>>>(dinv, N_AMINO);
    deg_scatter<<<ceilDiv(E_AMINO, 256), 256, 0, stream>>>(aei, dinv);
    deg_to_dinv<<<ceilDiv(N_AMINO, 256), 256, 0, stream>>>(dinv);
    edge_norm<<<ceilDiv(E_AMINO, 256), 256, 0, stream>>>(aei, dinv, nrm);

    // ================= ARMAConv: K=3 stacks, T=7 layers =================
    packA<<<ceilDiv(N_AMINO * 32, 256), 256, 0, stream>>>(
        aa, aabf, N_AMINO, F_AIN, F_AIN, 32);
    const long long NHK = (long long)KST * N_AMINO * H_ARMA;
    const dim3 ga(H_ARMA / 16, ceilDiv(N_AMINO / 16, 4));
    for (int t = 0; t < TT; ++t) {
        if (t == 0) {
            for (int k = 0; k < KST; ++k) {
                packT<<<ceilDiv(H_ARMA * 32, 256), 256, 0, stream>>>(
                    arma_init + (size_t)k * F_AIN * H_ARMA, btbuf,
                    F_AIN, H_ARMA, H_ARMA, 32, H_ARMA);
                launch_gemm(32, ga, wblk, stream, aabf, btbuf,
                            tmp + (size_t)k * N_AMINO * H_ARMA, H_ARMA, N_AMINO,
                            nullptr, 0);
            }
        } else {
            packA<<<ceilDiv(KST * N_AMINO * H_ARMA, 256), 256, 0, stream>>>(
                prev, Apack, KST * N_AMINO, H_ARMA, H_ARMA, H_ARMA);
            for (int k = 0; k < KST; ++k) {
                packT<<<ceilDiv(H_ARMA * H_ARMA, 256), 256, 0, stream>>>(
                    arma_w + ((size_t)(t - 1) * KST + k) * H_ARMA * H_ARMA, btbuf,
                    H_ARMA, H_ARMA, H_ARMA, H_ARMA, H_ARMA);
                launch_gemm(H_ARMA, ga, wblk, stream,
                            Apack + (size_t)k * N_AMINO * H_ARMA, btbuf,
                            tmp + (size_t)k * N_AMINO * H_ARMA, H_ARMA, N_AMINO,
                            nullptr, 0);
            }
        }
        for (int k = 0; k < KST; ++k) {
            // acc = aa @ root_w[t][k] + bias[t][k]  (messages added on top)
            packT<<<ceilDiv(H_ARMA * 32, 256), 256, 0, stream>>>(
                arma_root + ((size_t)t * KST + k) * F_AIN * H_ARMA, btbuf,
                F_AIN, H_ARMA, H_ARMA, 32, H_ARMA);
            launch_gemm(32, ga, wblk, stream, aabf, btbuf,
                        acc + (size_t)k * N_AMINO * H_ARMA, H_ARMA, N_AMINO,
                        arma_bias + ((size_t)t * KST + k) * H_ARMA, 1);
        }
        arma_scatter<<<ceilDiv((int)(KST * E_AMINO * H_ARMA), 256), 256, 0,
                       stream>>>(aei, nrm, tmp, acc);
        relu_copy<<<ceilDiv((int)NHK, 256), 256, 0, stream>>>(acc, prev, NHK);
    }
    arma_mean_relu<<<ceilDiv(N_AMINO * H_ARMA, 256), 256, 0, stream>>>(prev, aaout);

    // ================= per-graph pooling + MLP head =================
    zero_f32<<<64, 256, 0, stream>>>(g, (long long)BGRAPH * H_ARMA);
    pool_graph<<<ceilDiv(N_AMINO * H_ARMA, 256), 256, 0, stream>>>(abatch, aaout, g);

    packA<<<ceilDiv(BGRAPH * H_ARMA, 256), 256, 0, stream>>>(
        g, Apack, BGRAPH, H_ARMA, H_ARMA, H_ARMA);
    packT<<<ceilDiv(F1 * H_ARMA, 256), 256, 0, stream>>>(
        l1_w, btbuf, H_ARMA, F1, F1, H_ARMA, F1);
    launch_gemm(H_ARMA, dim3(F1 / 16, BGRAPH / 64), wblk, stream,
                Apack, btbuf, p1, F1, BGRAPH, l1_b, 2);

    packA<<<ceilDiv(BGRAPH * F1, 256), 256, 0, stream>>>(
        p1, Apack, BGRAPH, F1, F1, F1);
    packT<<<ceilDiv(F2 * F1, 256), 256, 0, stream>>>(
        l2_w, btbuf, F1, F2, F2, F1, F2);
    launch_gemm(F1, dim3(F2 / 16, BGRAPH / 64), wblk, stream,
                Apack, btbuf, p2, F2, BGRAPH, l2_b, 2);

    packA<<<ceilDiv(BGRAPH * F2, 256), 256, 0, stream>>>(
        p2, Apack, BGRAPH, F2, F2, F2);
    packT<<<ceilDiv(F3 * F2, 256), 256, 0, stream>>>(
        l3_w, btbuf, F2, F3, F3, F2, F3);
    launch_gemm(F2, dim3(F3 / 16, BGRAPH / 64), wblk, stream,
                Apack, btbuf, p3, F3, BGRAPH, l3_b, 2);

    final_dot<<<ceilDiv(BGRAPH, 256), 256, 0, stream>>>(p3, l4_w, l4_b, out);
    (void)in_sizes; (void)n_in; (void)out_size; (void)ws_size;
}
